// M6_GraphBiLSTM_16716012716044
// MI455X (gfx1250) — compile-verified, matrix operand order checked by reading, not run on hardware
//
#include <hip/hip_runtime.h>
#include <hip/hip_bf16.h>
#include <math.h>

typedef __attribute__((ext_vector_type(16))) _Float16 v16h_t;
typedef __attribute__((ext_vector_type(8)))  _Float16 v8h_t;
typedef __attribute__((ext_vector_type(8)))  float    v8f_t;

// ---------------------------------------------------------------------------
// WMMA fragment helpers (CDNA5 wave32 layouts, CDNA5 ISA §7.12.2)
//  A 16x32 f16: lanes 0-15 -> rows, K 0..7 + 16..23 ; lanes 16-31 -> K 8..15 + 24..31
//  The same layout serves B stored [N][K] ("NT"); all GEMMs here are NT after
//  one-off weight / V transposes, so every fragment is two aligned 16-B loads.
// ---------------------------------------------------------------------------
__device__ __forceinline__ v8f_t zero8() {
  v8f_t z;
#pragma unroll
  for (int i = 0; i < 8; ++i) z[i] = 0.0f;
  return z;
}

__device__ __forceinline__ v16h_t frag_rowmajor_f16(const _Float16* __restrict__ tile,
                                                    long long ld, int lane) {
  const _Float16* p = tile + (long long)(lane & 15) * ld + ((lane >> 4) << 3);
  v8h_t lo = *(const v8h_t*)(p);
  v8h_t hi = *(const v8h_t*)(p + 16);
  v16h_t r;
#pragma unroll
  for (int i = 0; i < 8; ++i) { r[i] = lo[i]; r[8 + i] = hi[i]; }
  return r;
}

__device__ __forceinline__ v16h_t frag_rowmajor_f32(const float* __restrict__ tile,
                                                    long long ld, int lane) {
  const float* p = tile + (long long)(lane & 15) * ld + ((lane >> 4) << 3);
  v16h_t r;
#pragma unroll
  for (int i = 0; i < 8; ++i) { r[i] = (_Float16)p[i]; r[8 + i] = (_Float16)p[16 + i]; }
  return r;
}

// ---------------------------------------------------------------------------
// Generic batched NT WMMA GEMM:
//   C[z] = act( A[z] (MxK, row-major) * B[z] ([N][K]) + bias[zo] + Cadd[z] )
//  block = 256 threads = 8 waves (4x2), block tile 128x64, wave tile 32x32.
//  Requires M == 128*gridDim.y, N == 64*gridDim.x, K % 32 == 0.
//  z = blockIdx.z, zo = z / ZI, zi = z % ZI (two-level batch strides).
// ---------------------------------------------------------------------------
template <bool AF32, bool OUT16, bool RELU>
__global__ void k_gemm(const void* __restrict__ Ap, const _Float16* __restrict__ Bp,
                       void* __restrict__ Cp, const float* __restrict__ Cadd,
                       const float* __restrict__ bias,
                       int K, int lda, int ldb, int ldc, int ldadd, int ZI,
                       long long sAo, long long sAi, long long sBo, long long sBi,
                       long long sCo, long long sCi, long long sDo, long long sDi,
                       long long sFo, long long dBase) {
  int z = blockIdx.z;
  int zo = z / ZI, zi = z - zo * ZI;
  long long aOff = (long long)zo * sAo + (long long)zi * sAi;
  long long bOff = (long long)zo * sBo + (long long)zi * sBi;
  long long cOff = (long long)zo * sCo + (long long)zi * sCi;
  long long dOff = dBase + (long long)zo * sDo + (long long)zi * sDi;
  long long fOff = (long long)zo * sFo;

  int lane = threadIdx.x & 31;
  int wv = threadIdx.x >> 5;
  int wm = wv & 3, wn = wv >> 2;
  long long row0 = (long long)blockIdx.y * 128 + wm * 32;
  int col0 = blockIdx.x * 64 + wn * 32;

  v8f_t acc00 = zero8(), acc01 = zero8(), acc10 = zero8(), acc11 = zero8();

  const char* Abytes = (const char*)Ap;
  for (int kc = 0; kc < K; kc += 32) {
    // prefetch the next A k-chunk for this lane (lowers to global_prefetch_b8)
    if (kc + 32 < K) {
      long long pfe = aOff + (row0 + (lane & 15)) * (long long)lda + kc + 32;
      __builtin_prefetch(Abytes + pfe * (AF32 ? 4 : 2), 0, 1);
    }
    v16h_t a0, a1;
    if (AF32) {
      const float* A = ((const float*)Ap) + aOff;
      a0 = frag_rowmajor_f32(A + row0 * lda + kc, lda, lane);
      a1 = frag_rowmajor_f32(A + (row0 + 16) * lda + kc, lda, lane);
    } else {
      const _Float16* A = ((const _Float16*)Ap) + aOff;
      a0 = frag_rowmajor_f16(A + row0 * lda + kc, lda, lane);
      a1 = frag_rowmajor_f16(A + (row0 + 16) * lda + kc, lda, lane);
    }
    const _Float16* Bt = Bp + bOff;
    v16h_t b0 = frag_rowmajor_f16(Bt + (long long)col0 * ldb + kc, ldb, lane);
    v16h_t b1 = frag_rowmajor_f16(Bt + (long long)(col0 + 16) * ldb + kc, ldb, lane);

    acc00 = __builtin_amdgcn_wmma_f32_16x16x32_f16(false, a0, false, b0, (short)0, acc00, false, false);
    acc01 = __builtin_amdgcn_wmma_f32_16x16x32_f16(false, a0, false, b1, (short)0, acc01, false, false);
    acc10 = __builtin_amdgcn_wmma_f32_16x16x32_f16(false, a1, false, b0, (short)0, acc10, false, false);
    acc11 = __builtin_amdgcn_wmma_f32_16x16x32_f16(false, a1, false, b1, (short)0, acc11, false, false);
  }

  int cr = (lane >> 4) << 3;  // C/D: lanes 16-31 hold rows v+8
  int cc = lane & 15;
#pragma unroll
  for (int mi = 0; mi < 2; ++mi)
#pragma unroll
    for (int ni = 0; ni < 2; ++ni) {
      v8f_t acc = (mi == 0) ? ((ni == 0) ? acc00 : acc01) : ((ni == 0) ? acc10 : acc11);
#pragma unroll
      for (int v = 0; v < 8; ++v) {
        long long r = row0 + mi * 16 + v + cr;
        int c2 = col0 + ni * 16 + cc;
        float x = acc[v];
        if (bias) x += bias[fOff + c2];
        if (Cadd) x += Cadd[dOff + r * (long long)ldadd + c2];
        if (RELU) x = fmaxf(x, 0.0f);
        if (OUT16) ((_Float16*)Cp)[cOff + r * (long long)ldc + c2] = (_Float16)x;
        else       ((float*)Cp)[cOff + r * (long long)ldc + c2] = x;
      }
    }
}

// ---------------------------------------------------------------------------
// Fused conv1(k7,p3)+BN+GELU -> conv2(k5,s2,p2)+BN+GELU, one (b,c) row / block.
// Writes feat[(b*128+t)][c][64] as f16 (GAT GEMM input layout).
// ---------------------------------------------------------------------------
__global__ void k_fused_conv(const float* __restrict__ x,
                             const float* __restrict__ w1, const float* __restrict__ b1,
                             const float* __restrict__ bn1g, const float* __restrict__ bn1b,
                             const float* __restrict__ w2, const float* __restrict__ b2,
                             const float* __restrict__ bn2g, const float* __restrict__ bn2b,
                             _Float16* __restrict__ feat) {
  __shared__ float xs[262];
  __shared__ float y1[32][260];
  __shared__ float w1s[32 * 7];
  int bc = blockIdx.x;  // 0..4095
  int b = bc >> 4, c = bc & 15;
  int tid = threadIdx.x;  // 256
  for (int i = tid; i < 32 * 7; i += 256) w1s[i] = w1[i];
  for (int i = tid; i < 262; i += 256) xs[i] = 0.0f;
  if (tid < 128) {  // zero conv2 halo columns of y1
    int oc = tid >> 2, p = tid & 3;
    y1[oc][p < 2 ? p : 256 + p] = 0.0f;
  }
  __syncthreads();
  const float* xr = x + (size_t)bc * 256;
  for (int t = tid; t < 256; t += 256) xs[3 + t] = xr[t];
  __syncthreads();
  const float invs = rsqrtf(1.0f + 1e-5f);
  for (int i = tid; i < 32 * 256; i += 256) {
    int oc = i >> 8, t = i & 255;
    float acc = b1[oc];
    const float* wk = &w1s[oc * 7];
#pragma unroll
    for (int k = 0; k < 7; ++k) acc += xs[t + k] * wk[k];
    acc = acc * (bn1g[oc] * invs) + bn1b[oc];
    acc = 0.5f * acc * (1.0f + erff(acc * 0.70710678118f));
    y1[oc][2 + t] = acc;
  }
  __syncthreads();
  for (int i = tid; i < 64 * 128; i += 256) {
    int oc = i >> 7, to = i & 127;
    float acc = b2[oc];
    const float* wk = w2 + oc * 160;  // [64][32][5], L2-resident
    for (int ic = 0; ic < 32; ++ic) {
#pragma unroll
      for (int k = 0; k < 5; ++k) acc += y1[ic][2 * to + k] * wk[ic * 5 + k];
    }
    acc = acc * (bn2g[oc] * invs) + bn2b[oc];
    acc = 0.5f * acc * (1.0f + erff(acc * 0.70710678118f));
    feat[((size_t)(b * 128 + to) * 16 + c) * 64 + oc] = (_Float16)acc;
  }
}

// ---------------------------------------------------------------------------
// Fused GAT attention + ReLU + LayerNorm, one graph (16 nodes x 128 feats)/block.
// h f16 [M][16][128] -> out f16 [M][16][128]
// ---------------------------------------------------------------------------
__global__ void k_gat_attn(const _Float16* __restrict__ h,
                           const float* __restrict__ asrc, const float* __restrict__ adst,
                           const float* __restrict__ gam, const float* __restrict__ bet,
                           _Float16* __restrict__ out) {
  __shared__ float hs[16 * 128];
  __shared__ float es[64], ed[64];
  __shared__ float attn[16 * 64];
  __shared__ float ov[16 * 128];
  __shared__ float red[256], red2[256];
  __shared__ float mu[16], iv[16];
  size_t m = blockIdx.x;
  int tid = threadIdx.x;  // 256
  const _Float16* hm = h + m * 2048;
  for (int i = tid; i < 2048; i += 256) hs[i] = (float)hm[i];
  __syncthreads();
  if (tid < 64) {
    int n = tid >> 2, hh = tid & 3;
    float s = 0.0f, d = 0.0f;
    const float* hp = &hs[n * 128 + hh * 32];
    for (int k = 0; k < 32; ++k) { s += hp[k] * asrc[hh * 32 + k]; d += hp[k] * adst[hh * 32 + k]; }
    es[tid] = s; ed[tid] = d;
  }
  __syncthreads();
  if (tid < 64) {
    int i = tid >> 2, hh = tid & 3;
    float e[16], mx = -1e30f;
    for (int j = 0; j < 16; ++j) {
      float v = es[i * 4 + hh] + ed[j * 4 + hh];
      v = (v > 0.0f) ? v : 0.2f * v;  // leaky relu 0.2
      e[j] = v; mx = fmaxf(mx, v);
    }
    float sum = 0.0f;
    for (int j = 0; j < 16; ++j) { e[j] = __expf(e[j] - mx); sum += e[j]; }
    float inv = 1.0f / sum;
    for (int j = 0; j < 16; ++j) attn[i * 64 + j * 4 + hh] = e[j] * inv;
  }
  __syncthreads();
  for (int idx = tid; idx < 2048; idx += 256) {
    int i = idx >> 7, dcol = idx & 127, hh = dcol >> 5;
    float acc = 0.0f;
    for (int j = 0; j < 16; ++j) acc += attn[i * 64 + j * 4 + hh] * hs[j * 128 + dcol];
    ov[idx] = fmaxf(acc, 0.0f);
  }
  __syncthreads();
  {  // layernorm per node over 128 feats: 16 threads per row
    int row = tid >> 4, sub = tid & 15;
    float ps = 0.0f, ps2 = 0.0f;
    for (int k = sub; k < 128; k += 16) { float v = ov[row * 128 + k]; ps += v; ps2 += v * v; }
    red[tid] = ps; red2[tid] = ps2;
    __syncthreads();
    if (tid < 16) {
      float s = 0.0f, s2 = 0.0f;
      for (int k = 0; k < 16; ++k) { s += red[tid * 16 + k]; s2 += red2[tid * 16 + k]; }
      float mean = s * (1.0f / 128.0f);
      float var = s2 * (1.0f / 128.0f) - mean * mean;
      mu[tid] = mean; iv[tid] = rsqrtf(var + 1e-5f);
    }
    __syncthreads();
  }
  _Float16* om = out + m * 2048;
  for (int idx = tid; idx < 2048; idx += 256) {
    int i = idx >> 7, dcol = idx & 127;
    om[idx] = (_Float16)((ov[idx] - mu[i]) * iv[i] * gam[dcol] + bet[dcol]);
  }
}

// ---------------------------------------------------------------------------
// LSTM cell (phase B): gates -> c,h update, writes hs (f16 + optional f32).
// gbuf [2][256][512], states [2][256][128], hs layout [b][t][dir*128+j].
// ---------------------------------------------------------------------------
__global__ void k_lstm_cell(const float* __restrict__ gbuf,
                            float* __restrict__ hst, float* __restrict__ cst,
                            _Float16* __restrict__ hs16, float* __restrict__ hs32, int s) {
  int idx = blockIdx.x * 256 + threadIdx.x;  // 65536
  if (idx >= 65536) return;
  int j = idx & 127;
  int b = (idx >> 7) & 255;
  int dir = idx >> 15;
  int te = dir ? (127 - s) : s;
  const float* g = gbuf + ((size_t)dir * 256 + b) * 512;
  float gi = g[j], gf = g[128 + j], gg = g[256 + j], go = g[384 + j];
  size_t sidx = ((size_t)dir * 256 + b) * 128 + j;
  float c = cst[sidx];
  float si = 1.0f / (1.0f + __expf(-gi));
  float sf = 1.0f / (1.0f + __expf(-gf));
  float so = 1.0f / (1.0f + __expf(-go));
  c = sf * c + si * tanhf(gg);
  float hv = so * tanhf(c);
  cst[sidx] = c; hst[sidx] = hv;
  size_t oidx = ((size_t)b * 128 + te) * 256 + dir * 128 + j;
  hs16[oidx] = (_Float16)hv;
  if (hs32) hs32[oidx] = hv;
}

// ---------------------------------------------------------------------------
// Softmax over rows of length 128 (with scale), f32 -> f16.
// ---------------------------------------------------------------------------
__global__ void k_softmax128(const float* __restrict__ in, _Float16* __restrict__ out, float scale) {
  __shared__ float red[128];
  size_t r = blockIdx.x;
  int t = threadIdx.x;  // 128
  float v = in[r * 128 + t] * scale;
  red[t] = v; __syncthreads();
  for (int o = 64; o > 0; o >>= 1) { if (t < o) red[t] = fmaxf(red[t], red[t + o]); __syncthreads(); }
  float mx = red[0]; __syncthreads();
  float e = __expf(v - mx);
  red[t] = e; __syncthreads();
  for (int o = 64; o > 0; o >>= 1) { if (t < o) red[t] += red[t + o]; __syncthreads(); }
  out[r * 128 + t] = (_Float16)(e / red[0]);
}

// ---------------------------------------------------------------------------
// Generic LayerNorm over D<=512, optional residual, f32/f16 outputs.
// ---------------------------------------------------------------------------
__global__ void k_layernorm(const float* __restrict__ X, const float* __restrict__ Rres,
                            const float* __restrict__ g, const float* __restrict__ bb,
                            float* __restrict__ out32, _Float16* __restrict__ out16, int D) {
  __shared__ float buf[512], red[256], red2[256];
  size_t r = blockIdx.x;
  int t = threadIdx.x;  // 256
  float ps = 0.0f, ps2 = 0.0f;
  for (int k = t; k < D; k += 256) {
    float v = X[r * D + k];
    if (Rres) v += Rres[r * D + k];
    buf[k] = v; ps += v; ps2 += v * v;
  }
  red[t] = ps; red2[t] = ps2; __syncthreads();
  for (int o = 128; o > 0; o >>= 1) { if (t < o) { red[t] += red[t + o]; red2[t] += red2[t + o]; } __syncthreads(); }
  float mean = red[0] / (float)D;
  float var = red2[0] / (float)D - mean * mean;
  float iv = rsqrtf(var + 1e-5f);
  for (int k = t; k < D; k += 256) {
    float v = (buf[k] - mean) * iv * g[k] + bb[k];
    if (out32) out32[r * D + k] = v;
    if (out16) out16[r * D + k] = (_Float16)v;
  }
}

// mean + max pooling over time: att [256][128][256] -> pooled [256][512]
__global__ void k_pool(const float* __restrict__ att, float* __restrict__ pooled) {
  int b = blockIdx.x, e = threadIdx.x;  // 256 threads
  const float* p = att + (size_t)b * 128 * 256 + e;
  float s = 0.0f, mx = -1e30f;
  for (int t = 0; t < 128; ++t) { float v = p[(size_t)t * 256]; s += v; mx = fmaxf(mx, v); }
  pooled[(size_t)b * 512 + e] = s * (1.0f / 128.0f);
  pooled[(size_t)b * 512 + 256 + e] = mx;
}

__global__ void k_cls2(const _Float16* __restrict__ h, const float* __restrict__ w,
                       const float* __restrict__ b2, float* __restrict__ out) {
  int b = blockIdx.x * blockDim.x + threadIdx.x;
  if (b >= 256) return;
  float acc = b2[0];
  for (int k = 0; k < 128; ++k) acc += (float)h[b * 128 + k] * w[k];
  out[b] = acc;
}

__global__ void k_cvt_f16(const float* __restrict__ in, _Float16* __restrict__ out, long long n) {
  long long i = (long long)blockIdx.x * blockDim.x + threadIdx.x;
  if (i < n) out[i] = (_Float16)in[i];
}

// in [R][C] f32 -> out [C][R] f16 (weight transpose to NT layout)
__global__ void k_transpose_cvt(const float* __restrict__ in, _Float16* __restrict__ out,
                                int R, int Ccols) {
  int idx = blockIdx.x * 256 + threadIdx.x;
  if (idx < R * Ccols) {
    int r = idx / Ccols, c = idx - r * Ccols;
    out[(size_t)c * R + r] = (_Float16)in[idx];
  }
}

// V slice of qkv [32768][768] -> vT [b][h][d=64][t=128] f16 (NT layout for attn@V)
__global__ void k_transpose_v(const _Float16* __restrict__ qkv, _Float16* __restrict__ vT) {
  int idx = blockIdx.x * 256 + threadIdx.x;  // 8388608
  if (idx >= 8388608) return;
  int t = idx & 127;
  int d = (idx >> 7) & 63;
  int h = (idx >> 13) & 3;
  int b = idx >> 15;
  vT[idx] = qkv[(size_t)(b * 128 + t) * 768 + 512 + h * 64 + d];
}

__global__ void k_add2(const float* __restrict__ a, const float* __restrict__ b,
                       float* __restrict__ o, int n) {
  int i = blockIdx.x * blockDim.x + threadIdx.x;
  if (i < n) o[i] = a[i] + b[i];
}

__global__ void k_zero_f32(float* __restrict__ p, int n) {
  int i = blockIdx.x * blockDim.x + threadIdx.x;
  if (i < n) p[i] = 0.0f;
}

// ---------------------------------------------------------------------------
extern "C" void kernel_launch(void* const* d_in, const int* in_sizes, int n_in,
                              void* d_out, int out_size, void* d_ws, size_t ws_size,
                              hipStream_t stream) {
  const float* x        = (const float*)d_in[0];
  const float* conv1_w  = (const float*)d_in[1];
  const float* conv1_b  = (const float*)d_in[2];
  const float* bn1_g    = (const float*)d_in[3];
  const float* bn1_b    = (const float*)d_in[4];
  const float* conv2_w  = (const float*)d_in[5];
  const float* conv2_b  = (const float*)d_in[6];
  const float* bn2_g    = (const float*)d_in[7];
  const float* bn2_b    = (const float*)d_in[8];
  const float* gat1_w   = (const float*)d_in[9];
  const float* gat1_as  = (const float*)d_in[10];
  const float* gat1_ad  = (const float*)d_in[11];
  const float* gn1_g    = (const float*)d_in[12];
  const float* gn1_b    = (const float*)d_in[13];
  const float* gat2_w   = (const float*)d_in[14];
  const float* gat2_as  = (const float*)d_in[15];
  const float* gat2_ad  = (const float*)d_in[16];
  const float* gn2_g    = (const float*)d_in[17];
  const float* gn2_b    = (const float*)d_in[18];
  const float* l_wih[4] = {(const float*)d_in[19], (const float*)d_in[23],
                           (const float*)d_in[27], (const float*)d_in[31]};
  const float* l_whh[4] = {(const float*)d_in[20], (const float*)d_in[24],
                           (const float*)d_in[28], (const float*)d_in[32]};
  const float* l_bih[4] = {(const float*)d_in[21], (const float*)d_in[25],
                           (const float*)d_in[29], (const float*)d_in[33]};
  const float* l_bhh[4] = {(const float*)d_in[22], (const float*)d_in[26],
                           (const float*)d_in[30], (const float*)d_in[34]};
  const float* mha_in_w  = (const float*)d_in[35];
  const float* mha_in_b  = (const float*)d_in[36];
  const float* mha_out_w = (const float*)d_in[37];
  const float* mha_out_b = (const float*)d_in[38];
  const float* an_g = (const float*)d_in[39];
  const float* an_b = (const float*)d_in[40];
  const float* pn_g = (const float*)d_in[41];
  const float* pn_b = (const float*)d_in[42];
  const float* cls1_w = (const float*)d_in[43];
  const float* cls1_b = (const float*)d_in[44];
  const float* cls2_w = (const float*)d_in[45];
  const float* cls2_b = (const float*)d_in[46];
  float* out = (float*)d_out;

  // -------- workspace bump allocator (256B aligned) --------
  char* ws = (char*)d_ws;
  size_t off = 0;
  auto WA = [&](size_t bytes) -> char* {
    char* p = ws + off;
    off += (bytes + 255) & ~(size_t)255;
    return p;
  };
  _Float16* featF16 = (_Float16*)WA(67108864);    // [32768*16][64]
  _Float16* hF16    = (_Float16*)WA(134217728);   // GAT h; reused as LSTM pre (f32)
  _Float16* g1F16   = (_Float16*)WA(134217728);   // GAT1 out; reused as MHA scratch
  _Float16* g2F16   = (_Float16*)WA(134217728);   // graph_out f16; reused as vT
  _Float16* gat1wT  = (_Float16*)WA(16384);       // [128][64]
  _Float16* gat2wT  = (_Float16*)WA(32768);       // [128][128]
  _Float16* wih0_16 = (_Float16*)WA(4194304);     // [2][512][2048]
  _Float16* whh0_16 = (_Float16*)WA(262144);      // [2][512][128]
  float*    bias0   = (float*)WA(4096);           // [2][512]
  _Float16* wih1_16 = (_Float16*)WA(524288);      // [2][512][256]
  _Float16* whh1_16 = (_Float16*)WA(262144);
  float*    bias1   = (float*)WA(4096);
  _Float16* mhaInW16  = (_Float16*)WA(393216);    // [768][256]
  _Float16* mhaOutW16 = (_Float16*)WA(131072);    // [256][256]
  _Float16* cls1w16   = (_Float16*)WA(131072);    // [128][512]
  float*    gbuf   = (float*)WA(2097152);         // [2][256][512]
  float*    hstate = (float*)WA(262144);          // [2][256][128]
  float*    cstate = (float*)WA(262144);
  _Float16* hsl0F16     = (_Float16*)WA(16777216);  // [256][128][256]
  _Float16* lstmOutF16  = (_Float16*)WA(16777216);
  float*    lstmOutF32  = (float*)WA(33554432);
  float*    attF32      = (float*)WA(33554432);
  float*    pooledF32   = (float*)WA(524288);
  _Float16* pooledF16   = (_Float16*)WA(262144);
  _Float16* clsH16      = (_Float16*)WA(65536);
  // region reuse (lifetimes disjoint):
  float*    pre     = (float*)hF16;                          // [2][32768][512] f32
  float*    scores  = (float*)featF16;                       // [1024][128][128] f32
  _Float16* attn16  = (_Float16*)g1F16;                      // 32 MB
  _Float16* o16     = (_Float16*)((char*)g1F16 + 33554432);  // 16 MB
  float*    mhaOut  = (float*)((char*)g1F16 + 50331648);     // 32 MB
  _Float16* qkv16   = (_Float16*)((char*)g1F16 + 83886080);  // 48 MB
  _Float16* vT16    = (_Float16*)g2F16;                      // [1024][64][128] (graph_out retired)
  (void)ws_size; (void)n_in; (void)in_sizes; (void)out_size;

  dim3 blk(256);
  const long long Z0 = 0;

  // 1) fused conv stack -> featF16
  k_fused_conv<<<4096, blk, 0, stream>>>(x, conv1_w, conv1_b, bn1_g, bn1_b,
                                         conv2_w, conv2_b, bn2_g, bn2_b, featF16);

  // 2) weight conversions (all weights end up in NT [N][K] f16 layout)
  k_transpose_cvt<<<32, blk, 0, stream>>>(gat1_w, gat1wT, 64, 128);
  k_transpose_cvt<<<64, blk, 0, stream>>>(gat2_w, gat2wT, 128, 128);
  for (int d = 0; d < 2; ++d) {
    k_cvt_f16<<<4096, blk, 0, stream>>>(l_wih[d], wih0_16 + (size_t)d * 1048576, 1048576);
    k_cvt_f16<<<256, blk, 0, stream>>>(l_whh[d], whh0_16 + (size_t)d * 65536, 65536);
    k_add2<<<2, blk, 0, stream>>>(l_bih[d], l_bhh[d], bias0 + d * 512, 512);
    k_cvt_f16<<<512, blk, 0, stream>>>(l_wih[2 + d], wih1_16 + (size_t)d * 131072, 131072);
    k_cvt_f16<<<256, blk, 0, stream>>>(l_whh[2 + d], whh1_16 + (size_t)d * 65536, 65536);
    k_add2<<<2, blk, 0, stream>>>(l_bih[2 + d], l_bhh[2 + d], bias1 + d * 512, 512);
  }
  k_cvt_f16<<<768, blk, 0, stream>>>(mha_in_w, mhaInW16, 196608);
  k_cvt_f16<<<256, blk, 0, stream>>>(mha_out_w, mhaOutW16, 65536);
  k_cvt_f16<<<256, blk, 0, stream>>>(cls1_w, cls1w16, 65536);

  // 3) GAT1: h = feat @ W1  (M=524288,K=64,N=128) -> hF16
  k_gemm<false, true, false><<<dim3(2, 4096, 1), blk, 0, stream>>>(
      featF16, gat1wT, hF16, nullptr, nullptr, 64, 64, 64, 128, 0, 1,
      Z0, Z0, Z0, Z0, Z0, Z0, Z0, Z0, Z0, Z0);
  k_gat_attn<<<32768, blk, 0, stream>>>(hF16, gat1_as, gat1_ad, gn1_g, gn1_b, g1F16);
  // GAT2
  k_gemm<false, true, false><<<dim3(2, 4096, 1), blk, 0, stream>>>(
      g1F16, gat2wT, hF16, nullptr, nullptr, 128, 128, 128, 128, 0, 1,
      Z0, Z0, Z0, Z0, Z0, Z0, Z0, Z0, Z0, Z0);
  k_gat_attn<<<32768, blk, 0, stream>>>(hF16, gat2_as, gat2_ad, gn2_g, gn2_b, g2F16);
  // g2F16 == graph_out [32768][2048]; hF16 region now free -> pre

  // 4) LSTM layer 0 input projection: pre[dir] = graph_out @ wih^T + (bih+bhh)
  k_gemm<false, false, false><<<dim3(8, 256, 2), blk, 0, stream>>>(
      g2F16, wih0_16, pre, nullptr, bias0, 2048, 2048, 2048, 512, 0, 1,
      Z0, Z0, (long long)512 * 2048, Z0, 16777216LL, Z0, Z0, Z0, 512LL, Z0);
  // recurrence
  k_zero_f32<<<256, blk, 0, stream>>>(hstate, 65536);
  k_zero_f32<<<256, blk, 0, stream>>>(cstate, 65536);
  for (int s = 0; s < 128; ++s) {
    long long sDo = 16777216LL + (long long)(127 - 2 * s) * 512;
    k_gemm<true, false, false><<<dim3(8, 2, 2), blk, 0, stream>>>(
        hstate, whh0_16, gbuf, pre, nullptr, 128, 128, 128, 512, 65536, 1,
        32768LL, Z0, 65536LL, Z0, 131072LL, Z0, sDo, Z0, Z0, (long long)s * 512);
    k_lstm_cell<<<256, blk, 0, stream>>>(gbuf, hstate, cstate, hsl0F16, nullptr, s);
  }

  // 5) LSTM layer 1
  k_gemm<false, false, false><<<dim3(8, 256, 2), blk, 0, stream>>>(
      hsl0F16, wih1_16, pre, nullptr, bias1, 256, 256, 256, 512, 0, 1,
      Z0, Z0, 131072LL, Z0, 16777216LL, Z0, Z0, Z0, 512LL, Z0);
  k_zero_f32<<<256, blk, 0, stream>>>(hstate, 65536);
  k_zero_f32<<<256, blk, 0, stream>>>(cstate, 65536);
  for (int s = 0; s < 128; ++s) {
    long long sDo = 16777216LL + (long long)(127 - 2 * s) * 512;
    k_gemm<true, false, false><<<dim3(8, 2, 2), blk, 0, stream>>>(
        hstate, whh1_16, gbuf, pre, nullptr, 128, 128, 128, 512, 65536, 1,
        32768LL, Z0, 65536LL, Z0, 131072LL, Z0, sDo, Z0, Z0, (long long)s * 512);
    k_lstm_cell<<<256, blk, 0, stream>>>(gbuf, hstate, cstate, lstmOutF16, lstmOutF32, s);
  }

  // 6) MHA. qkv = lstm_out @ in_w^T + in_b  (M=32768,K=256,N=768) -> f16
  k_gemm<false, true, false><<<dim3(12, 256, 1), blk, 0, stream>>>(
      lstmOutF16, mhaInW16, qkv16, nullptr, mha_in_b, 256, 256, 256, 768, 0, 1,
      Z0, Z0, Z0, Z0, Z0, Z0, Z0, Z0, Z0, Z0);
  // transpose V into NT layout [b][h][d][t]
  k_transpose_v<<<32768, blk, 0, stream>>>(qkv16, vT16);
  // scores[b,h] = q @ k^T  (batched z=1024, ZI=4: zo=b, zi=h)
  k_gemm<false, false, false><<<dim3(2, 1, 1024), blk, 0, stream>>>(
      qkv16, qkv16 + 256, scores, nullptr, nullptr, 64, 768, 768, 128, 0, 4,
      98304LL, 64LL, 98304LL, 64LL, 65536LL, 16384LL, Z0, Z0, Z0, Z0);
  k_softmax128<<<131072, dim3(128), 0, stream>>>(scores, attn16, 0.125f);
  // o[b,t,h*64+d] = attn @ v  (B = vT [d][t], NT path)
  k_gemm<false, true, false><<<dim3(1, 1, 1024), blk, 0, stream>>>(
      attn16, vT16, o16, nullptr, nullptr, 128, 128, 128, 256, 0, 4,
      65536LL, 16384LL, 32768LL, 8192LL, 32768LL, 64LL, Z0, Z0, Z0, Z0);
  // out projection
  k_gemm<false, false, false><<<dim3(4, 256, 1), blk, 0, stream>>>(
      o16, mhaOutW16, mhaOut, nullptr, mha_out_b, 256, 256, 256, 256, 0, 1,
      Z0, Z0, Z0, Z0, Z0, Z0, Z0, Z0, Z0, Z0);
  // residual + layernorm
  k_layernorm<<<32768, blk, 0, stream>>>(mhaOut, lstmOutF32, an_g, an_b, attF32, nullptr, 256);

  // 7) pooling + norm + classifier
  k_pool<<<256, blk, 0, stream>>>(attF32, pooledF32);
  k_layernorm<<<256, blk, 0, stream>>>(pooledF32, nullptr, pn_g, pn_b, nullptr, pooledF16, 512);
  k_gemm<false, true, true><<<dim3(2, 2, 1), blk, 0, stream>>>(
      pooledF16, cls1w16, clsH16, nullptr, cls1_b, 512, 512, 512, 128, 0, 1,
      Z0, Z0, Z0, Z0, Z0, Z0, Z0, Z0, Z0, Z0);
  k_cls2<<<1, blk, 0, stream>>>(clsH16, cls2_w, cls2_b, out);
}
